// LogitsProcessorWithTopping_63814624084201
// MI455X (gfx1250) — compile-verified, matrix operand order checked by reading, not run on hardware
//
#include <hip/hip_runtime.h>
#include <stdint.h>

// Problem sizes (fixed by the reference)
#define B_TOK 256
#define H_DIM 2048
#define V_DIM 32000
#define N_GRP 5          // base head + 4 adapters

// CDNA5 WMMA vector types (wave32)
typedef __attribute__((ext_vector_type(16))) __bf16         v16bf;
typedef __attribute__((ext_vector_type(8)))  float          v8f;
typedef __attribute__((ext_vector_type(4)))  unsigned short us4;

// fp32 -> bf16, round-to-nearest-even
__device__ __forceinline__ unsigned short f32_to_bf16(float f) {
  unsigned int u = __float_as_uint(f);
  u += 0x7FFFu + ((u >> 16) & 1u);
  return (unsigned short)(u >> 16);
}

// ---------------------------------------------------------------------------
// Kernel 1: compact tokens by group. cnt[g], perm[g*256 + j] = token id.
// ---------------------------------------------------------------------------
__global__ __launch_bounds__(256) void compact_kernel(
    const int* __restrict__ widx, int* __restrict__ cnt, int* __restrict__ perm) {
  int t = threadIdx.x;
  if (t < N_GRP) cnt[t] = 0;
  __syncthreads();
  if (t < B_TOK) {
    int g = widx[t] + 1;                 // -1 (base) -> 0
    int p = atomicAdd(&cnt[g], 1);
    perm[g * 256 + p] = t;
  }
}

// ---------------------------------------------------------------------------
// Kernel 2: gather + convert hidden_states rows to compacted bf16 X per group.
// One block per (g, j) slot; 256 threads copy a 2048-elem row (fully coalesced).
// ---------------------------------------------------------------------------
__global__ __launch_bounds__(256) void gather_kernel(
    const float* __restrict__ x, const int* __restrict__ cnt,
    const int* __restrict__ perm, unsigned short* __restrict__ Xg) {
  int g = blockIdx.x >> 8;
  int j = blockIdx.x & 255;
  if (j >= cnt[g]) return;
  int tokenId = perm[g * 256 + j];
  const float* src = x + (size_t)tokenId * H_DIM;
  unsigned short* dst = Xg + ((size_t)(g * 256 + j)) * H_DIM;
  int t = threadIdx.x;
#pragma unroll
  for (int i = 0; i < 2; ++i) {
    int base = t * 8 + i * 4;
    float4 f = *(const float4*)(src + base);
    us4 o;
    o.x = f32_to_bf16(f.x); o.y = f32_to_bf16(f.y);
    o.z = f32_to_bf16(f.z); o.w = f32_to_bf16(f.w);
    *(us4*)(dst + base) = o;
  }
}

// ---------------------------------------------------------------------------
// Kernel 3: grouped GEMM via V_WMMA_F32_16X16X32_BF16.
// grid = (V/128, N_GRP, 4 token-supertiles of 64). Block = 256 threads (8 waves).
// Wave (wr, wc): 16 tokens x 64 vocab cols -> 4 f32 accumulator fragments.
// K is processed in chunks of 64 with a DOUBLE-BUFFERED LDS stage of the
// 64(K) x 128(N) weight chunk as bf16 (n-major, stride 72 shorts = 144 B so
// fragment b128 reads stay 16B-aligned and banks are staggered).
// One __syncthreads per 8 WMMAs.
// ---------------------------------------------------------------------------
#define BS_STRIDE 72     // shorts per n-row (64 K values + 8 pad)

__global__ __launch_bounds__(256) void grouped_lmhead_gemm(
    const float* __restrict__ weight,        // [H, V]  (base head, h-major rows)
    const float* __restrict__ delta,         // [N_ADAPT, V, H] (v-major rows)
    const unsigned short* __restrict__ Xg,   // compacted bf16 [N_GRP*256, H]
    const int* __restrict__ cnt,
    const int* __restrict__ perm,
    float* __restrict__ out)                 // [B, V]
{
  const int vt = blockIdx.x;                 // vocab tile (128 cols)
  const int g  = blockIdx.y;                 // group
  const int mt = blockIdx.z;                 // token supertile (64 rows)
  const int c_g = cnt[g];
  if (mt * 64 >= c_g) return;                // uniform branch: EXEC stays full

  const int v0   = vt * 128;
  const int tid  = threadIdx.x;
  const int lane = tid & 31;
  const int wid  = tid >> 5;
  const int wr   = wid & 3;                  // token sub-tile within supertile
  const int wc   = wid >> 2;                 // 64-col half
  const int rbase = mt * 64 + wr * 16;
  const int cbase = wc * 64;

  const int laneHalf = lane >> 4;
  const int lmod     = lane & 15;
  const int ko       = laneHalf * 8;         // K offset per 16-bit WMMA layout

  __shared__ unsigned short Bs[2][128 * BS_STRIDE];   // ~36 KB double buffer

  // Per-lane output token ids (C/D layout: M = r + laneHalf*8, N = lane&15)
  int tok[8];
#pragma unroll
  for (int r = 0; r < 8; ++r) {
    int rowg = rbase + r + laneHalf * 8;
    tok[r] = (rowg < c_g) ? perm[g * 256 + rowg] : -1;
  }

  v8f acc[4];
#pragma unroll
  for (int c = 0; c < 4; ++c)
    acc[c] = (v8f){0.f, 0.f, 0.f, 0.f, 0.f, 0.f, 0.f, 0.f};

  const float* Wbase = (g == 0) ? weight
                                : (delta + (size_t)(g - 1) * V_DIM * H_DIM);
  // A-matrix row for this lane (M = lane&15), bf16, contiguous in K
  const unsigned short* Arow =
      Xg + ((size_t)(g * 256 + rbase + lmod)) * H_DIM;

  // Staging thread mappings (computed once)
  const int a_tn  = tid >> 4;                // adapter: n-row 0..15 (+16*jj)
  const int a_tk4 = (tid & 15) * 4;          // adapter: k within chunk, 0..60
  const int b_tk  = tid >> 5;                // base: k-row 0..7 (+8*jj)
  const int b_tn4 = tid & 31;                // base: n = b_tn4*4

  for (int k0 = 0, it = 0; k0 < H_DIM; k0 += 64, ++it) {
    unsigned short* bs = Bs[it & 1];

    // ---- Stage 64(K) x 128(N) weight chunk: all loads first, then convert ---
    float4 w[8];
    if (g == 0) {
      // base head: weight[k][v], rows contiguous in v -> coalesced float4 in n
#pragma unroll
      for (int jj = 0; jj < 8; ++jj) {
        int k = b_tk + jj * 8;
        w[jj] = *(const float4*)(Wbase + (size_t)(k0 + k) * V_DIM + v0 + b_tn4 * 4);
      }
#pragma unroll
      for (int jj = 0; jj < 8; ++jj) {
        int k = b_tk + jj * 8;
        int n = b_tn4 * 4;
        bs[(n + 0) * BS_STRIDE + k] = f32_to_bf16(w[jj].x);
        bs[(n + 1) * BS_STRIDE + k] = f32_to_bf16(w[jj].y);
        bs[(n + 2) * BS_STRIDE + k] = f32_to_bf16(w[jj].z);
        bs[(n + 3) * BS_STRIDE + k] = f32_to_bf16(w[jj].w);
      }
    } else {
      // adapters: delta[v][h], rows contiguous in h -> coalesced float4 in k
#pragma unroll
      for (int jj = 0; jj < 8; ++jj) {
        int n = a_tn + jj * 16;
        w[jj] = *(const float4*)(Wbase + (size_t)(v0 + n) * H_DIM + k0 + a_tk4);
      }
      // speculative prefetch of next K chunk (dropped silently if OOB)
      __builtin_prefetch(Wbase + (size_t)(v0 + a_tn) * H_DIM + k0 + 64 + a_tk4, 0, 1);
#pragma unroll
      for (int jj = 0; jj < 8; ++jj) {
        int n = a_tn + jj * 16;
        us4 o;
        o.x = f32_to_bf16(w[jj].x); o.y = f32_to_bf16(w[jj].y);
        o.z = f32_to_bf16(w[jj].z); o.w = f32_to_bf16(w[jj].w);
        *(us4*)(&bs[n * BS_STRIDE + a_tk4]) = o;
      }
    }
    __syncthreads();   // producers done; consumers of other buffer already past

    // ---- A fragments: 16x32 bf16 each, two K sub-chunks ---------------------
    // Lane: M=lane&15; slots 0..7 = K ko..ko+7, slots 8..15 = K 16+ko..16+ko+7
    union { uint4 u[2]; v16bf v; } afrag[2];
#pragma unroll
    for (int s = 0; s < 2; ++s) {
      afrag[s].u[0] = *(const uint4*)(Arow + k0 + s * 32 + ko);
      afrag[s].u[1] = *(const uint4*)(Arow + k0 + s * 32 + ko + 16);
    }

    // ---- 8 WMMAs per chunk (4 col tiles x 2 K sub-chunks) -------------------
#pragma unroll
    for (int c = 0; c < 4; ++c) {
      const unsigned short* bp0 =
          bs + (cbase + c * 16 + lmod) * BS_STRIDE + ko;
#pragma unroll
      for (int s = 0; s < 2; ++s) {
        union { uint4 u[2]; v16bf v; } bfrag;
        bfrag.u[0] = *(const uint4*)(bp0 + s * 32);
        bfrag.u[1] = *(const uint4*)(bp0 + s * 32 + 16);
        acc[c] = __builtin_amdgcn_wmma_f32_16x16x32_bf16(
            false, afrag[s].v, false, bfrag.v, (short)0, acc[c], false, false);
      }
    }
  }

  // Scatter results by real token id (C/D layout: M = r + laneHalf*8, N = lmod)
#pragma unroll
  for (int c = 0; c < 4; ++c) {
    int col = v0 + cbase + c * 16 + lmod;
#pragma unroll
    for (int r = 0; r < 8; ++r) {
      if (tok[r] >= 0) out[(size_t)tok[r] * V_DIM + col] = acc[c][r];
    }
  }
}

// ---------------------------------------------------------------------------
extern "C" void kernel_launch(void* const* d_in, const int* in_sizes, int n_in,
                              void* d_out, int out_size, void* d_ws, size_t ws_size,
                              hipStream_t stream) {
  const float* hidden = (const float*)d_in[0];   // [B, H] f32
  const float* weight = (const float*)d_in[1];   // [H, V] f32
  const float* delta  = (const float*)d_in[2];   // [N_ADAPT, V, H] f32
  const int*   widx   = (const int*)d_in[3];     // [B] i32 in [-1, N_ADAPT)
  float* out = (float*)d_out;                    // [B, V] f32

  // Workspace layout (~5.25 MB):
  //   [0,32)      : int cnt[N_GRP] (padded)
  //   [32,5152)   : int perm[N_GRP*256]
  //   [8192, ...) : bf16 Xg[N_GRP*256, H]
  char* ws = (char*)d_ws;
  int* cnt  = (int*)ws;
  int* perm = (int*)(ws + 32);
  unsigned short* Xg = (unsigned short*)(ws + 8192);

  compact_kernel<<<1, 256, 0, stream>>>(widx, cnt, perm);
  gather_kernel<<<dim3(N_GRP * 256), 256, 0, stream>>>(hidden, cnt, perm, Xg);
  grouped_lmhead_gemm<<<dim3(V_DIM / 128, N_GRP, 4), 256, 0, stream>>>(
      weight, delta, Xg, cnt, perm, out);
}
